// Hecto_60851096649855
// MI455X (gfx1250) — matrix operand order repaired and checked by reading, not compiled, not run on hardware
//
#include <hip/hip_runtime.h>
#include <hip/hip_bf16.h>
#include <math.h>

typedef __attribute__((ext_vector_type(16))) _Float16 v16h;
typedef __attribute__((ext_vector_type(8)))  _Float16 v8h;
typedef __attribute__((ext_vector_type(8)))  float    v8f;

#if defined(__has_builtin)
#if __has_builtin(__builtin_amdgcn_global_load_async_to_lds_b128) && \
    __has_builtin(__builtin_amdgcn_s_wait_asynccnt)
#define USE_ASYNC_LDS 1
#endif
#endif

namespace {
constexpr int B    = 4;
constexpr int Q    = 300;
constexpr int D    = 256;
constexpr int H    = 512;
constexpr int NP   = 6;
constexpr int SMAX = NP * Q;      // 1800
constexpr int NQL  = 16;
constexpr int T    = NQL + SMAX;  // 1816
constexpr int SCTX = 256;
constexpr int DEPTHL = 4;
constexpr int HEADS  = 8;
constexpr int HD     = 64;
constexpr int FF     = 2048;
constexpr int NCLS   = 396;
constexpr int NCTX   = 64;
constexpr int NCLASSES = 91;
}

#if defined(USE_ASYNC_LDS)
// 16-byte payload type matching the builtin's expected pointee
typedef int a_i4 __attribute__((vector_size(16)));
// global generic pointer -> addrspace(1) int4*
__device__ __forceinline__ __attribute__((address_space(1))) a_i4*
glob_i4(const float* p) {
  return (__attribute__((address_space(1))) a_i4*)(unsigned long long)p;
}
// generic LDS pointer -> addrspace(3) int4* (LDS byte offset = low 32 bits)
__device__ __forceinline__ __attribute__((address_space(3))) a_i4*
lds_i4(float* p) {
  return (__attribute__((address_space(3))) a_i4*)(unsigned int)(unsigned long long)p;
}
#endif

// ---------------------------------------------------------------------------
// Generic WMMA GEMM: C[M,N] = act(A[M,K] @ W[K,N] + bias), f32 in, f16 WMMA
// operands, f32 accumulate.  Requires K % 32 == 0 and N % 64 == 0.
// Block: 256 threads / 8 waves, tile 64(M) x 64(N); wave = (mi 0..3, ni 0..1),
// each wave owns one A fragment and two 16-wide N sub-tiles (2 WMMA / K-step).
// A/B tiles staged in LDS as raw f32 (async global->LDS when available).
// amode: 0 row-major (lda=K); 1 column-major A[k*lda+r]; 2 im2col conv1d
//        (stride2,k3,pad1): k->(t=k>>9,ci=k&511), li=2r-1+t, A[li*512+ci].
// act: 0 none, 1 relu, 2 exact gelu.  rowLen: zero rows with
// (gm % rowMod) >= rowLen[gm / rowMod].
// ---------------------------------------------------------------------------
struct GemmArgs {
  const float* A; const float* W; const float* bias; float* C;
  int M, N, K, lda;
  long strideA, strideC;   // per-batch (blockIdx.z) element strides
  int amode, act;
  const int* rowLen; int rowMod;
};

__global__ __launch_bounds__(256) void k_gemm(GemmArgs g) {
  __shared__ float As[64][36];   // [m][k], +4 pad (row 144B, 16B-aligned)
  __shared__ float Bs[32][68];   // [k][n], +4 pad (row 272B, 16B-aligned)
  const int tid  = threadIdx.x;
  const int lane = tid & 31;
  const int wave = tid >> 5;
  const int hgrp = lane >> 4;
  const int lm   = lane & 15;
  const int mi   = wave & 3;        // 4 M sub-tiles of 16
  const int ni   = wave >> 2;       // 2 N groups of 32
  const int m0   = blockIdx.y * 64;
  const int n0   = blockIdx.x * 64;
  const float* A = g.A + (long)blockIdx.z * g.strideA;
  float*       C = g.C + (long)blockIdx.z * g.strideC;
#if defined(USE_ASYNC_LDS)
  const bool asyncA = (g.amode == 0) && (m0 + 64 <= g.M);
#endif

  v8f acc0 = {}, acc1 = {};
  for (int k0 = 0; k0 < g.K; k0 += 32) {
    // ---- stage A tile 64x32 f32 (8 floats / thread) ----
#if defined(USE_ASYNC_LDS)
    if (asyncA) {
      #pragma unroll
      for (int it = 0; it < 2; it++) {
        const int f = tid * 4 + it * 1024;
        const int r = f >> 5, k = f & 31;
        __builtin_amdgcn_global_load_async_to_lds_b128(
            glob_i4(A + (long)(m0 + r) * g.lda + k0 + k),
            lds_i4(&As[r][k]), 0, 0);
      }
    } else
#endif
    {
      #pragma unroll
      for (int it = 0; it < 2; it++) {
        const int f = tid * 4 + it * 1024;
        const int r = f >> 5, k = f & 31;
        const int gm = m0 + r;
        float vals[4];
        if (g.amode == 0) {
          if (gm < g.M) {
            const float4 f4 = *(const float4*)(A + (long)gm * g.lda + k0 + k);
            vals[0] = f4.x; vals[1] = f4.y; vals[2] = f4.z; vals[3] = f4.w;
          } else {
            vals[0] = vals[1] = vals[2] = vals[3] = 0.f;
          }
        } else if (g.amode == 1) {
          const int  gmc = (gm < g.M) ? gm : 0;
          const float sc = (gm < g.M) ? 1.f : 0.f;
          #pragma unroll
          for (int i = 0; i < 4; i++)
            vals[i] = A[(long)(k0 + k + i) * g.lda + gmc] * sc;
        } else {
          #pragma unroll
          for (int i = 0; i < 4; i++) {
            const int kk = k0 + k + i;
            const int t  = kk >> 9, ci = kk & 511;
            const int li = 2 * gm - 1 + t;
            const bool ok = (gm < g.M) && (li >= 0) && (li < g.lda);
            const int lic = ok ? li : 0;
            vals[i] = A[(long)lic * 512 + ci] * (ok ? 1.f : 0.f);
          }
        }
        #pragma unroll
        for (int i = 0; i < 4; i++) As[r][k + i] = vals[i];
      }
    }
    // ---- stage B tile 32x64 f32 (8 floats / thread) ----
    {
      #pragma unroll
      for (int it = 0; it < 2; it++) {
        const int f = tid * 4 + it * 1024;
        const int k = f >> 6, n = f & 63;
        const float* Wp = g.W + (long)(k0 + k) * g.N + n0 + n;
#if defined(USE_ASYNC_LDS)
        __builtin_amdgcn_global_load_async_to_lds_b128(
            glob_i4(Wp), lds_i4(&Bs[k][n]), 0, 0);
#else
        const float4 f4 = *(const float4*)Wp;
        Bs[k][n + 0] = f4.x; Bs[k][n + 1] = f4.y;
        Bs[k][n + 2] = f4.z; Bs[k][n + 3] = f4.w;
#endif
      }
    }
    if (k0 + 32 < g.K) {
      __builtin_prefetch(g.W + (long)(k0 + 32) * g.N + n0, 0, 1);
    }
#if defined(USE_ASYNC_LDS)
    __builtin_amdgcn_s_wait_asynccnt(0);
#endif
    __syncthreads();

    // ---- A fragment (ISA 7.12.2 16-bit A layout) ----
    const int mrow = mi * 16 + lm;
    const int ab = hgrp ? 8 : 0;          // K = ab+j (j<8), ab+16+(j-8)
    const float4 alo0 = *(const float4*)&As[mrow][ab];
    const float4 alo1 = *(const float4*)&As[mrow][ab + 4];
    const float4 ahi0 = *(const float4*)&As[mrow][ab + 16];
    const float4 ahi1 = *(const float4*)&As[mrow][ab + 20];
    v16h a;
    a[0]  = (_Float16)alo0.x; a[1]  = (_Float16)alo0.y;
    a[2]  = (_Float16)alo0.z; a[3]  = (_Float16)alo0.w;
    a[4]  = (_Float16)alo1.x; a[5]  = (_Float16)alo1.y;
    a[6]  = (_Float16)alo1.z; a[7]  = (_Float16)alo1.w;
    a[8]  = (_Float16)ahi0.x; a[9]  = (_Float16)ahi0.y;
    a[10] = (_Float16)ahi0.z; a[11] = (_Float16)ahi0.w;
    a[12] = (_Float16)ahi1.x; a[13] = (_Float16)ahi1.y;
    a[14] = (_Float16)ahi1.z; a[15] = (_Float16)ahi1.w;

    // ---- B fragments: lane column n, K = c0+j (j<8), c0+16+(j-8) ----
    const int c0 = hgrp ? 8 : 0;
    const int nA = ni * 32 + lm;
    const int nBcol = ni * 32 + 16 + lm;
    v16h b0m, b1m;
    #pragma unroll
    for (int j = 0; j < 16; j++) {
      b0m[j] = (_Float16)Bs[c0 + j + (j & 8)][nA];
      b1m[j] = (_Float16)Bs[c0 + j + (j & 8)][nBcol];
    }
    acc0 = __builtin_amdgcn_wmma_f32_16x16x32_f16(false, a, false, b0m,
                                                  (short)0, acc0, false, false);
    acc1 = __builtin_amdgcn_wmma_f32_16x16x32_f16(false, a, false, b1m,
                                                  (short)0, acc1, false, false);
    __syncthreads();
  }
  // ---- epilogue ----
  const int gn0 = n0 + ni * 32 + lm;
  const int gn1 = gn0 + 16;
  const float bv0 = g.bias ? g.bias[gn0] : 0.f;
  const float bv1 = g.bias ? g.bias[gn1] : 0.f;
  #pragma unroll
  for (int r = 0; r < 8; r++) {
    const int gm = m0 + mi * 16 + r + 8 * hgrp;
    if (gm < g.M) {
      float v0 = acc0[r] + bv0;
      float v1 = acc1[r] + bv1;
      if (g.act == 1) {
        v0 = fmaxf(v0, 0.f); v1 = fmaxf(v1, 0.f);
      } else if (g.act == 2) {
        v0 = 0.5f * v0 * (1.f + erff(v0 * 0.70710678118f));
        v1 = 0.5f * v1 * (1.f + erff(v1 * 0.70710678118f));
      }
      if (g.rowLen) {
        const int bb = gm / g.rowMod, ss = gm % g.rowMod;
        if (ss >= g.rowLen[bb]) { v0 = 0.f; v1 = 0.f; }
      }
      C[(long)gm * g.N + gn0] = v0;
      C[(long)gm * g.N + gn1] = v1;
    }
  }
}

// ---------------------------------------------------------------------------
// Flash attention, one wave per (16-query tile, head, batch).
// Online softmax over 32-key chunks; S = Q K^T via 2 chained WMMAs,
// O += P V via 4 WMMAs; P converted C-layout -> A-layout through LDS.
// ---------------------------------------------------------------------------
__global__ __launch_bounds__(32) void k_attn(const float* Qm, const float* Km,
                                             const float* Vm,
                                             const unsigned char* mk,
                                             float* Om, int Tq, int Sk) {
  const int lane = threadIdx.x;
  const int hgrp = lane >> 4, lm = lane & 15;
  const int m0 = blockIdx.x * 16;
  const int h  = blockIdx.y;
  const int b  = blockIdx.z;
  __shared__ _Float16 P[16][40];

  const float* Qb = Qm + ((long)b * Tq) * H + h * HD;
  const float* Kb = Km + ((long)b * Sk) * H + h * HD;
  const float* Vb = Vm + ((long)b * Sk) * H + h * HD;
  const unsigned char* mb = mk ? mk + (long)b * Sk : nullptr;

  v16h qa[2];
  {
    const int m = m0 + lm;
    const bool ok = m < Tq;
    const float* qr = Qb + (long)m * H;
    #pragma unroll
    for (int c = 0; c < 2; c++) {
      const int base0 = c * 32 + (hgrp ? 8 : 0);
      #pragma unroll
      for (int j = 0; j < 8; j++) {
        qa[c][j]     = (_Float16)(ok ? qr[base0 + j] : 0.f);
        qa[c][j + 8] = (_Float16)(ok ? qr[base0 + 16 + j] : 0.f);
      }
    }
  }

  float rm[8], rl[8];
  #pragma unroll
  for (int r = 0; r < 8; r++) { rm[r] = -1e30f; rl[r] = 0.f; }
  v8f accO[4] = {};

  const int nch = (Sk + 31) / 32;
  for (int kc = 0; kc < nch; kc++) {
    const int u0 = kc * 32;
    v8f s[2];
    bool vok[2];
    #pragma unroll
    for (int t = 0; t < 2; t++) {
      const int u  = u0 + t * 16 + lm;
      const bool ok = (u < Sk) && (!mb || mb[u] == 0);
      vok[t] = ok;
      const int us = (u < Sk) ? u : 0;
      const float* kr = Kb + (long)us * H;
      v8f c = {};
      #pragma unroll
      for (int cch = 0; cch < 2; cch++) {
        const int kb0 = cch * 32 + (hgrp ? 8 : 0);
        v16h kv;
        #pragma unroll
        for (int j = 0; j < 16; j++) kv[j] = (_Float16)kr[kb0 + j + (j & 8)];
        c = __builtin_amdgcn_wmma_f32_16x16x32_f16(false, qa[cch], false, kv,
                                                   (short)0, c, false, false);
      }
      s[t] = c;
    }
    float af[8];
    #pragma unroll
    for (int r = 0; r < 8; r++) {
      const float v0 = vok[0] ? (s[0][r] * 0.125f) : -1e30f;
      const float v1 = vok[1] ? (s[1][r] * 0.125f) : -1e30f;
      float mx = fmaxf(v0, v1);
      #pragma unroll
      for (int off = 1; off < 16; off <<= 1)
        mx = fmaxf(mx, __shfl_xor(mx, off, 32));
      const float mnew = fmaxf(rm[r], mx);
      const float p0 = vok[0] ? __expf(v0 - mnew) : 0.f;
      const float p1 = vok[1] ? __expf(v1 - mnew) : 0.f;
      float ps = p0 + p1;
      #pragma unroll
      for (int off = 1; off < 16; off <<= 1)
        ps += __shfl_xor(ps, off, 32);
      af[r] = __expf(rm[r] - mnew);
      rl[r] = rl[r] * af[r] + ps;
      rm[r] = mnew;
      P[r + 8 * hgrp][lm]      = (_Float16)p0;
      P[r + 8 * hgrp][16 + lm] = (_Float16)p1;
    }
    #pragma unroll
    for (int nt = 0; nt < 4; nt++)
      #pragma unroll
      for (int r = 0; r < 8; r++) accO[nt][r] *= af[r];
    __syncthreads();
    v16h pa;
    {
      const int pb0 = hgrp ? 8 : 0;
      v8h lo = *(const v8h*)&P[lm][pb0];
      v8h hi = *(const v8h*)&P[lm][pb0 + 16];
      #pragma unroll
      for (int i = 0; i < 8; i++) { pa[i] = lo[i]; pa[i + 8] = hi[i]; }
    }
    #pragma unroll
    for (int nt = 0; nt < 4; nt++) {
      const int cc = nt * 16 + lm;
      v16h vv;
      #pragma unroll
      for (int j = 0; j < 16; j++) {
        const int kl = (hgrp ? 8 : 0) + j + (j & 8);
        const int u  = u0 + kl;
        const int us = (u < Sk) ? u : 0;
        vv[j] = (_Float16)((u < Sk) ? Vb[(long)us * H + cc] : 0.f);
      }
      accO[nt] = __builtin_amdgcn_wmma_f32_16x16x32_f16(false, pa, false, vv,
                                                        (short)0, accO[nt],
                                                        false, false);
    }
    __syncthreads();
  }
  #pragma unroll
  for (int nt = 0; nt < 4; nt++) {
    const int cc = nt * 16 + lm;
    #pragma unroll
    for (int r = 0; r < 8; r++) {
      const int m = m0 + r + 8 * hgrp;
      if (m < Tq)
        Om[((long)b * Tq + m) * H + h * HD + cc] =
            accO[nt][r] / fmaxf(rl[r], 1e-20f);
    }
  }
}

// ---------------------------------------------------------------------------
__global__ void k_select(const float* logits, const float* hs, float* selected,
                         int* lengths, unsigned char* maskB) {
  const int b = blockIdx.x, tid = threadIdx.x;
  __shared__ int idxmap[SMAX];
  __shared__ int cnt_s;
  for (int s = tid; s < SMAX; s += blockDim.x) {
    const int c = s / Q, q = s % Q;
    const float lg = logits[((long)(b * Q + q)) * NCLASSES + c];
    const float sg = 1.f / (1.f + __expf(-lg));
    idxmap[s] = (sg > 0.2f) ? 1 : 0;
  }
  __syncthreads();
  if (tid == 0) {
    int c = 0;
    for (int s = 0; s < SMAX; s++) {
      if (idxmap[s]) idxmap[s] = c++; else idxmap[s] = -1;
    }
    if (c == 0) { idxmap[0] = 0; c = 1; }   // fallback: hs[b,0]
    cnt_s = c; lengths[b] = c;
  }
  __syncthreads();
  const int cnt = cnt_s;
  for (int e = cnt * D + tid; e < SMAX * D; e += blockDim.x)
    selected[((long)b * SMAX) * D + e] = 0.f;
  for (int e = tid; e < SMAX * D; e += blockDim.x) {
    const int s = e / D, d = e % D;
    const int im = idxmap[s];
    if (im >= 0)
      selected[((long)(b * SMAX + im)) * D + d] =
          hs[((long)(b * Q + s % Q)) * D + d];
  }
  for (int t = tid; t < T; t += blockDim.x)
    maskB[(long)b * T + t] = (t >= NQL && (t - NQL) >= cnt) ? 1 : 0;
}

__global__ void k_assemble_x(const float* lq, const float* selproj, float* x) {
  const long i = (long)blockIdx.x * blockDim.x + threadIdx.x;
  if (i >= (long)B * T * H) return;
  const int hcol = (int)(i % H);
  const long row = i / H;
  const int t = (int)(row % T), b = (int)(row / T);
  x[i] = (t < NQL) ? lq[t * H + hcol]
                   : selproj[((long)(b * SMAX + (t - NQL))) * H + hcol];
}

// Wconv[co,ci,t] -> Wk[(t*512+ci)*512 + co]
__global__ void k_repack_convw(const float* Wc, float* Wk) {
  const int i = blockIdx.x * blockDim.x + threadIdx.x;
  if (i >= 512 * 512 * 3) return;
  const int co = i % 512;
  const int ci = (i / 512) % 512;
  const int t  = i / (512 * 512);
  Wk[i] = Wc[(co * 512 + ci) * 3 + t];
}

__global__ void k_pool_ctx(const float* cin, float* ctx, int Lc, int lvl) {
  const int n = blockIdx.x;
  const int b = blockIdx.y;
  const int start = (n * Lc) / NCTX;
  const int end   = ((n + 1) * Lc + NCTX - 1) / NCTX;
  const float inv = 1.f / (float)(end - start);
  for (int co = threadIdx.x; co < H; co += blockDim.x) {
    float acc = 0.f;
    for (int l = start; l < end; l++)
      acc += cin[((long)b * Lc + l) * H + co];
    ctx[((long)(b * SCTX + lvl * NCTX + n)) * H + co] = acc * inv;
  }
}

__global__ __launch_bounds__(256) void k_add_ln(float* x, const float* y,
                                                const float* g, const float* bt,
                                                int rows) {
  const int row = blockIdx.x;
  if (row >= rows) return;
  const int tid = threadIdx.x;
  __shared__ float red[256];
  const long base = (long)row * H;
  const float v0 = x[base + tid] + y[base + tid];
  const float v1 = x[base + tid + 256] + y[base + tid + 256];
  red[tid] = v0 + v1;
  __syncthreads();
  for (int o = 128; o > 0; o >>= 1) {
    if (tid < o) red[tid] += red[tid + o];
    __syncthreads();
  }
  const float mean = red[0] / (float)H;
  __syncthreads();
  const float d0 = v0 - mean, d1 = v1 - mean;
  red[tid] = d0 * d0 + d1 * d1;
  __syncthreads();
  for (int o = 128; o > 0; o >>= 1) {
    if (tid < o) red[tid] += red[tid + o];
    __syncthreads();
  }
  const float rstd = rsqrtf(red[0] / (float)H + 1e-5f);
  x[base + tid]       = d0 * rstd * g[tid] + bt[tid];
  x[base + tid + 256] = d1 * rstd * g[tid + 256] + bt[tid + 256];
}

__global__ void k_maskpool(const float* x, const unsigned char* mk,
                           const int* len, float* pooled) {
  const int b = blockIdx.x;
  const float denom = (float)(NQL + len[b]);
  for (int ch = threadIdx.x; ch < H; ch += blockDim.x) {
    float acc = 0.f;
    for (int t = 0; t < T; t++)
      if (!mk[(long)b * T + t]) acc += x[((long)(b * T + t)) * H + ch];
    pooled[b * H + ch] = acc / denom;
  }
}

__global__ __launch_bounds__(512) void k_head(const float* pooled,
                                              const float* W1, const float* b1,
                                              const float* W2, const float* b2,
                                              float* out) {
  const int b = blockIdx.x, j = threadIdx.x;
  __shared__ float hsh[H];
  float a = b1[j];
  for (int i = 0; i < H; i++) a += pooled[b * H + i] * W1[i * H + j];
  hsh[j] = fmaxf(a, 0.f);
  __syncthreads();
  if (j < NCLS) {
    float o = b2[j];
    for (int i = 0; i < H; i++) o += hsh[i] * W2[i * NCLS + j];
    out[b * NCLS + j] = o;
  }
}

// ---------------------------------------------------------------------------
extern "C" void kernel_launch(void* const* d_in, const int* in_sizes, int n_in,
                              void* d_out, int out_size, void* d_ws,
                              size_t ws_size, hipStream_t stream) {
  (void)in_sizes; (void)n_in; (void)out_size; (void)ws_size;
  const float* pred  = (const float*)d_in[0];
  const float* hs    = (const float*)d_in[1];
  const float* feat[4] = {(const float*)d_in[2], (const float*)d_in[3],
                          (const float*)d_in[4], (const float*)d_in[5]};
  const float* lq    = (const float*)d_in[6];
  const float* Wqp   = (const float*)d_in[7];
  const float* bqp   = (const float*)d_in[8];
  const float* Wip   = (const float*)d_in[9];
  const float* bip   = (const float*)d_in[10];
  const float* Wconv = (const float*)d_in[11];
  const float* bconv = (const float*)d_in[12];
  const float* saW   = (const float*)d_in[13];
  const float* sab   = (const float*)d_in[14];
  const float* caW   = (const float*)d_in[15];
  const float* cab   = (const float*)d_in[16];
  const float* ffW1  = (const float*)d_in[17];
  const float* ffb1  = (const float*)d_in[18];
  const float* ffW2  = (const float*)d_in[19];
  const float* ffb2  = (const float*)d_in[20];
  const float* lng   = (const float*)d_in[21];
  const float* lnb   = (const float*)d_in[22];
  const float* cW1   = (const float*)d_in[23];
  const float* cb1   = (const float*)d_in[24];
  const float* cW2   = (const float*)d_in[25];
  const float* cb2   = (const float*)d_in[26];

  float* ws = (float*)d_ws;
  long off = 0;
  auto alloc = [&](long n) { float* p = ws + off; off += n; return p; };
  float* x    = alloc((long)B * T * H);
  float* qb   = alloc((long)B * T * H);
  float* kb   = alloc((long)B * T * H);
  float* vb   = alloc((long)B * T * H);
  float* t0   = alloc((long)B * T * H);
  float* big0 = alloc(20480000L);        // proj / selected / FFN hidden
  float* big1 = alloc(10240000L);        // conv out / selproj
  float* ctx  = alloc((long)B * SCTX * H);
  float* wk   = alloc(1536L * 512);
  float* pooled = alloc(B * H);
  int* lengths = (int*)alloc(64);
  unsigned char* maskB = (unsigned char*)alloc(((long)B * T) / 4 + 64);

  auto gemm = [&](const float* A, const float* W, const float* bias, float* C,
                  int M, int N, int K, int lda, long sA, long sC, int nb,
                  int amode, int act, const int* rlen, int rmod) {
    GemmArgs ga;
    ga.A = A; ga.W = W; ga.bias = bias; ga.C = C;
    ga.M = M; ga.N = N; ga.K = K; ga.lda = lda;
    ga.strideA = sA; ga.strideC = sC;
    ga.amode = amode; ga.act = act; ga.rowLen = rlen; ga.rowMod = rmod;
    dim3 grid(N / 64, (M + 63) / 64, nb);
    hipLaunchKernelGGL(k_gemm, grid, dim3(256), 0, stream, ga);
  };

  hipLaunchKernelGGL(k_select, dim3(B), dim3(256), 0, stream,
                     pred, hs, big0, lengths, maskB);
  gemm(big0, Wqp, bqp, big1, B * SMAX, H, D, D, 0, 0, 1, 0, 0, lengths, SMAX);
  {
    const long nx = (long)B * T * H;
    hipLaunchKernelGGL(k_assemble_x, dim3((unsigned)((nx + 255) / 256)),
                       dim3(256), 0, stream, lq, big1, x);
  }
  hipLaunchKernelGGL(k_repack_convw, dim3((512 * 512 * 3 + 255) / 256),
                     dim3(256), 0, stream, Wconv, wk);
  const int HWs[4] = {10000, 2500, 625, 169};
  for (int lvl = 0; lvl < 4; lvl++) {
    const int HW = HWs[lvl];
    const int Lc = (HW + 1) / 2;
    gemm(feat[lvl], Wip, bip, big0, HW, H, D, HW,
         (long)D * HW, (long)HW * H, B, 1, 0, nullptr, 0);
    gemm(big0, wk, bconv, big1, Lc, H, 3 * 512, HW,
         (long)HW * H, (long)Lc * H, B, 2, 2, nullptr, 0);
    hipLaunchKernelGGL(k_pool_ctx, dim3(NCTX, B), dim3(256), 0, stream,
                       big1, ctx, Lc, lvl);
  }
  for (int l = 0; l < DEPTHL; l++) {
    const float* W  = saW + (long)l * 4 * H * H;
    const float* bb = sab + (long)l * 4 * H;
    gemm(x, W + 0L * H * H, bb + 0 * H, qb, B * T, H, H, H, 0, 0, 1, 0, 0, nullptr, 0);
    gemm(x, W + 1L * H * H, bb + 1 * H, kb, B * T, H, H, H, 0, 0, 1, 0, 0, nullptr, 0);
    gemm(x, W + 2L * H * H, bb + 2 * H, vb, B * T, H, H, H, 0, 0, 1, 0, 0, nullptr, 0);
    hipLaunchKernelGGL(k_attn, dim3((T + 15) / 16, HEADS, B), dim3(32), 0,
                       stream, qb, kb, vb, maskB, t0, T, T);
    gemm(t0, W + 3L * H * H, bb + 3 * H, qb, B * T, H, H, H, 0, 0, 1, 0, 0, nullptr, 0);
    hipLaunchKernelGGL(k_add_ln, dim3(B * T), dim3(256), 0, stream,
                       x, qb, lng + (l * 3 + 0) * H, lnb + (l * 3 + 0) * H, B * T);
    const float* Wc = caW + (long)l * 4 * H * H;
    const float* bc = cab + (long)l * 4 * H;
    gemm(x,   Wc + 0L * H * H, bc + 0 * H, qb, B * T,    H, H, H, 0, 0, 1, 0, 0, nullptr, 0);
    gemm(ctx, Wc + 1L * H * H, bc + 1 * H, kb, B * SCTX, H, H, H, 0, 0, 1, 0, 0, nullptr, 0);
    gemm(ctx, Wc + 2L * H * H, bc + 2 * H, vb, B * SCTX, H, H, H, 0, 0, 1, 0, 0, nullptr, 0);
    hipLaunchKernelGGL(k_attn, dim3((T + 15) / 16, HEADS, B), dim3(32), 0,
                       stream, qb, kb, vb, (const unsigned char*)nullptr, t0, T, SCTX);
    gemm(t0, Wc + 3L * H * H, bc + 3 * H, qb, B * T, H, H, H, 0, 0, 1, 0, 0, nullptr, 0);
    hipLaunchKernelGGL(k_add_ln, dim3(B * T), dim3(256), 0, stream,
                       x, qb, lng + (l * 3 + 1) * H, lnb + (l * 3 + 1) * H, B * T);
    gemm(x, ffW1 + (long)l * H * FF, ffb1 + l * FF, big0, B * T, FF, H, H,
         0, 0, 1, 0, 1, nullptr, 0);
    gemm(big0, ffW2 + (long)l * FF * H, ffb2 + l * H, t0, B * T, H, FF, FF,
         0, 0, 1, 0, 0, nullptr, 0);
    hipLaunchKernelGGL(k_add_ln, dim3(B * T), dim3(256), 0, stream,
                       x, t0, lng + (l * 3 + 2) * H, lnb + (l * 3 + 2) * H, B * T);
  }
  hipLaunchKernelGGL(k_maskpool, dim3(B), dim3(256), 0, stream,
                     x, maskB, lengths, pooled);
  hipLaunchKernelGGL(k_head, dim3(B), dim3(512), 0, stream,
                     pooled, cW1, cb1, cW2, cb2, (float*)d_out);
}
